// GAT_87076166959944
// MI455X (gfx1250) — matrix operand order repaired and checked by reading, not compiled, not run on hardware
//
#include <hip/hip_runtime.h>
#include <hip/hip_bf16.h>
#include <math.h>

// ---------------------------------------------------------------------------
// Types for CDNA5 WMMA: 16x16x32 bf16, f32 accumulate.
// ---------------------------------------------------------------------------
typedef __attribute__((ext_vector_type(16))) __bf16 v16bf;
typedef __attribute__((ext_vector_type(8)))  float  v8f;
typedef __attribute__((ext_vector_type(4)))  float  fl4;
typedef __attribute__((ext_vector_type(4)))  __bf16 bf4;

__device__ inline void atomicMaxF(float* addr, float v) {
  // monotone int/uint bit trick; valid for buffers initialized to -inf
  if (v >= 0.f) atomicMax((int*)addr, __float_as_int(v));
  else          atomicMin((unsigned int*)addr, __float_as_uint(v));
}

__device__ inline float lrelu(float x) { return x > 0.f ? x : 0.2f * x; }

__device__ inline void edge_sd(const int* __restrict__ ei, int e, int E,
                               int& s, int& d) {
  if (e < E) { s = ei[e]; d = ei[E + e]; }
  else       { s = e - E; d = s; }          // appended self-loops
}

// ---------------------------------------------------------------------------
// GEMM: C[M,NC] = A[M,K] @ B[K,NC], bf16 operands (converted on the fly via
// v_cvt_pk_bf16_f32), f32 accumulate via v_wmma_f32_16x16x32_bf16.
// One wave computes TWO 16x16 output tiles stacked in M (reuses B fragment).
// Requires K % 32 == 0, NC % 16 == 0. All index math fits in int32.
// ---------------------------------------------------------------------------
__device__ inline v16bf load_a_frag(const float* __restrict__ p) {
  // 16 bf16 from two 8-float contiguous groups: K {0..7} and K {16..23}
  const fl4* p4 = (const fl4*)p;
  bf4 c0 = __builtin_convertvector(p4[0], bf4);
  bf4 c1 = __builtin_convertvector(p4[1], bf4);
  bf4 c2 = __builtin_convertvector(p4[4], bf4);
  bf4 c3 = __builtin_convertvector(p4[5], bf4);
  v16bf a;
#pragma unroll
  for (int i = 0; i < 4; ++i) {
    a[i]      = c0[i];
    a[i + 4]  = c1[i];
    a[i + 8]  = c2[i];
    a[i + 12] = c3[i];
  }
  return a;
}

__global__ __launch_bounds__(32) void gemm_bf16_wmma(
    const float* __restrict__ A, const float* __restrict__ B,
    float* __restrict__ C, int M, int K, int NC, int tiles_n) {
  int tile = blockIdx.x;
  int tm   = (tile / tiles_n) * 32;     // first row of the 32-row super-tile
  int tn   = (tile % tiles_n) * 16;
  int lane = threadIdx.x & 31;
  int half = lane >> 4;                 // 0: lanes 0-15, 1: lanes 16-31
  int l15  = lane & 15;

  int row0 = tm + l15;                  // A row, tile 0
  int row1 = row0 + 16;                 // A row, tile 1
  int rA0  = row0 < M ? row0 : M - 1;
  int rA1  = row1 < M ? row1 : M - 1;
  int colB = tn + l15;

  v8f acc0 = {}, acc1 = {};
  for (int k0 = 0; k0 < K; k0 += 32) {
    // B 32x16: lane holds column colB; half 0 -> K {k0..k0+7, k0+16..k0+23},
    // half 1 -> +8; 2 packed bf16 per VGPR (same K order as A).
    v16bf b;
    const float* bcol = B + (k0 + half * 8) * NC + colB;
#pragma unroll
    for (int i = 0; i < 16; ++i) b[i] = (__bf16)bcol[(i + (i & 8)) * NC];

    // A 16x32: lane half 0 -> K {k0..k0+7, k0+16..k0+23}; half 1 -> +8
    int ka = k0 + half * 8;
    v16bf a0 = load_a_frag(A + rA0 * K + ka);
    v16bf a1 = load_a_frag(A + rA1 * K + ka);

    acc0 = __builtin_amdgcn_wmma_f32_16x16x32_bf16(
        false, a0, false, b, (short)0, acc0, false, false);
    acc1 = __builtin_amdgcn_wmma_f32_16x16x32_bf16(
        false, a1, false, b, (short)0, acc1, false, false);
  }

  // D layout: VGPR r -> row base + half*8 + r, col tn + l15.
  // Wave-uniform fast path: full 32-row super-tile -> unguarded stores.
  int srow = tm + half * 8;
  float* c0 = C + srow * NC + colB;
  if (tm + 32 <= M) {
#pragma unroll
    for (int r = 0; r < 8; ++r) {
      c0[r * NC]        = acc0[r];
      c0[(r + 16) * NC] = acc1[r];
    }
  } else {
#pragma unroll
    for (int r = 0; r < 8; ++r) {
      if (srow + r < M)      c0[r * NC]        = acc0[r];
      if (srow + r + 16 < M) c0[(r + 16) * NC] = acc1[r];
    }
  }
}

// ---------------------------------------------------------------------------
// Attention score precompute
// ---------------------------------------------------------------------------
__global__ void attn_scores8(const float* __restrict__ h,
                             const float* __restrict__ att_src,
                             const float* __restrict__ att_dst,
                             float* __restrict__ a_src,
                             float* __restrict__ a_dst, int N) {
  int idx = blockIdx.x * blockDim.x + threadIdx.x;   // node*8 + head
  if (idx >= N * 8) return;
  int n = idx >> 3, hd = idx & 7;
  const float* row = h + n * 64 + hd * 8;
  float s = 0.f, d = 0.f;
#pragma unroll
  for (int c = 0; c < 8; ++c) {
    float v = row[c];
    s += v * att_src[hd * 8 + c];
    d += v * att_dst[hd * 8 + c];
  }
  a_src[idx] = s;
  a_dst[idx] = d;
}

__global__ void attn_scores1(const float* __restrict__ g,
                             const float* __restrict__ att_src,
                             const float* __restrict__ att_dst,
                             float* __restrict__ a_src,
                             float* __restrict__ a_dst, int N) {
  int n = blockIdx.x * blockDim.x + threadIdx.x;
  if (n >= N) return;
  const float* row = g + n * 32;
  float s = 0.f, d = 0.f;
#pragma unroll
  for (int c = 0; c < 32; ++c) {
    float v = row[c];
    s += v * att_src[c];
    d += v * att_dst[c];
  }
  a_src[n] = s;
  a_dst[n] = d;
}

// ---------------------------------------------------------------------------
// Fills
// ---------------------------------------------------------------------------
__global__ void fill_f(float* __restrict__ p, float v, int n) {
  int i = blockIdx.x * blockDim.x + threadIdx.x;
  if (i < n) p[i] = v;
}
__global__ void fill_bias(float* __restrict__ p, const float* __restrict__ b,
                          int mask, int n) {
  int i = blockIdx.x * blockDim.x + threadIdx.x;
  if (i < n) p[i] = b[i & mask];
}

// ---------------------------------------------------------------------------
// Layer-1 edge passes (H = 8 heads, C = 8 channels)
// ---------------------------------------------------------------------------
__global__ void edge_max8(const int* __restrict__ ei,
                          const float* __restrict__ as,
                          const float* __restrict__ ad,
                          float* __restrict__ amax, int E, int ET) {
  int idx = blockIdx.x * blockDim.x + threadIdx.x;
  if (idx >= ET * 8) return;
  int e = idx >> 3, hd = idx & 7;
  int s, d; edge_sd(ei, e, E, s, d);
  float al = lrelu(as[s * 8 + hd] + ad[d * 8 + hd]);
  atomicMaxF(&amax[d * 8 + hd], al);
}

__global__ void edge_den8(const int* __restrict__ ei,
                          const float* __restrict__ as,
                          const float* __restrict__ ad,
                          const float* __restrict__ amax,
                          float* __restrict__ den, int E, int ET) {
  int idx = blockIdx.x * blockDim.x + threadIdx.x;
  if (idx >= ET * 8) return;
  int e = idx >> 3, hd = idx & 7;
  int s, d; edge_sd(ei, e, E, s, d);
  float al = lrelu(as[s * 8 + hd] + ad[d * 8 + hd]);
  float ex = __expf(al - amax[d * 8 + hd]);
  atomicAdd(&den[d * 8 + hd], ex);
}

__global__ void edge_agg8(const int* __restrict__ ei,
                          const float* __restrict__ as,
                          const float* __restrict__ ad,
                          const float* __restrict__ amax,
                          const float* __restrict__ den,
                          const float* __restrict__ h,
                          float* __restrict__ agg, int E, int ET) {
  int idx = blockIdx.x * blockDim.x + threadIdx.x;
  if (idx >= ET * 8) return;
  int e = idx >> 3, hd = idx & 7;
  int s, d; edge_sd(ei, e, E, s, d);
  float al = lrelu(as[s * 8 + hd] + ad[d * 8 + hd]);
  float w = __expf(al - amax[d * 8 + hd]) / (den[d * 8 + hd] + 1e-16f);
  const float* hs = h + s * 64 + hd * 8;
  float* od = agg + d * 64 + hd * 8;
#pragma unroll
  for (int c = 0; c < 8; ++c) atomicAdd(&od[c], hs[c] * w);
}

// ---------------------------------------------------------------------------
// ELU
// ---------------------------------------------------------------------------
__global__ void elu_k(const float* __restrict__ in, float* __restrict__ out,
                      int n) {
  int i = blockIdx.x * blockDim.x + threadIdx.x;
  if (i >= n) return;
  float v = in[i];
  out[i] = v > 0.f ? v : (__expf(v) - 1.f);
}

// ---------------------------------------------------------------------------
// Layer-2 edge passes (H = 1, C = 32)
// ---------------------------------------------------------------------------
__global__ void edge_max1(const int* __restrict__ ei,
                          const float* __restrict__ as,
                          const float* __restrict__ ad,
                          float* __restrict__ amax, int E, int ET) {
  int e = blockIdx.x * blockDim.x + threadIdx.x;
  if (e >= ET) return;
  int s, d; edge_sd(ei, e, E, s, d);
  atomicMaxF(&amax[d], lrelu(as[s] + ad[d]));
}

__global__ void edge_den1(const int* __restrict__ ei,
                          const float* __restrict__ as,
                          const float* __restrict__ ad,
                          const float* __restrict__ amax,
                          float* __restrict__ den, int E, int ET) {
  int e = blockIdx.x * blockDim.x + threadIdx.x;
  if (e >= ET) return;
  int s, d; edge_sd(ei, e, E, s, d);
  float ex = __expf(lrelu(as[s] + ad[d]) - amax[d]);
  atomicAdd(&den[d], ex);
}

__global__ void edge_agg1(const int* __restrict__ ei,
                          const float* __restrict__ as,
                          const float* __restrict__ ad,
                          const float* __restrict__ amax,
                          const float* __restrict__ den,
                          const float* __restrict__ g,
                          float* __restrict__ out, int E, int ET) {
  int idx = blockIdx.x * blockDim.x + threadIdx.x;   // edge*4 + group
  if (idx >= ET * 4) return;
  int e = idx >> 2, grp = idx & 3;
  int s, d; edge_sd(ei, e, E, s, d);
  float al = lrelu(as[s] + ad[d]);
  float w = __expf(al - amax[d]) / (den[d] + 1e-16f);
  const float* gs = g + s * 32 + grp * 8;
  float* od = out + d * 32 + grp * 8;
#pragma unroll
  for (int c = 0; c < 8; ++c) atomicAdd(&od[c], gs[c] * w);
}

// ---------------------------------------------------------------------------
// Launch
// ---------------------------------------------------------------------------
static inline int cdiv(long long a, long long b) { return (int)((a + b - 1) / b); }

extern "C" void kernel_launch(void* const* d_in, const int* in_sizes, int n_in,
                              void* d_out, int out_size, void* d_ws, size_t ws_size,
                              hipStream_t stream) {
  (void)n_in; (void)out_size; (void)ws_size;
  const float* x    = (const float*)d_in[0];
  const int*   ei   = (const int*)  d_in[1];
  const float* W1   = (const float*)d_in[2];
  const float* aS1w = (const float*)d_in[3];
  const float* aD1w = (const float*)d_in[4];
  const float* b1   = (const float*)d_in[5];
  const float* W2   = (const float*)d_in[6];
  const float* aS2w = (const float*)d_in[7];
  const float* aD2w = (const float*)d_in[8];
  const float* b2   = (const float*)d_in[9];
  float* out = (float*)d_out;

  const int N  = in_sizes[0] / 256;
  const int E  = in_sizes[1] / 2;
  const int ET = E + N;

  // Workspace layout (floats)
  float* f = (float*)d_ws;
  size_t o = 0;
  float* h1   = f + o; o += (size_t)N * 64;
  float* aS1  = f + o; o += (size_t)N * 8;
  float* aD1  = f + o; o += (size_t)N * 8;
  float* mx1  = f + o; o += (size_t)N * 8;
  float* dn1  = f + o; o += (size_t)N * 8;
  float* agg1 = f + o; o += (size_t)N * 64;
  float* h2   = f + o; o += (size_t)N * 64;
  float* g2   = f + o; o += (size_t)N * 32;
  float* aS2  = f + o; o += (size_t)N;
  float* aD2  = f + o; o += (size_t)N;
  float* mx2  = f + o; o += (size_t)N;
  float* dn2  = f + o; o += (size_t)N;

  const int TB = 256;

  // ---- Layer 1 ----
  {
    int tn = 64 / 16;
    gemm_bf16_wmma<<<cdiv(N, 32) * tn, 32, 0, stream>>>(x, W1, h1, N, 256, 64, tn);
  }
  attn_scores8<<<cdiv((long long)N * 8, TB), TB, 0, stream>>>(h1, aS1w, aD1w, aS1, aD1, N);
  fill_f<<<cdiv((long long)N * 8, TB), TB, 0, stream>>>(mx1, -INFINITY, N * 8);
  fill_f<<<cdiv((long long)N * 8, TB), TB, 0, stream>>>(dn1, 0.f, N * 8);
  fill_bias<<<cdiv((long long)N * 64, TB), TB, 0, stream>>>(agg1, b1, 63, N * 64);
  edge_max8<<<cdiv((long long)ET * 8, TB), TB, 0, stream>>>(ei, aS1, aD1, mx1, E, ET);
  edge_den8<<<cdiv((long long)ET * 8, TB), TB, 0, stream>>>(ei, aS1, aD1, mx1, dn1, E, ET);
  edge_agg8<<<cdiv((long long)ET * 8, TB), TB, 0, stream>>>(ei, aS1, aD1, mx1, dn1, h1, agg1, E, ET);
  elu_k<<<cdiv((long long)N * 64, TB), TB, 0, stream>>>(agg1, h2, N * 64);

  // ---- Layer 2 ----
  {
    int tn = 32 / 16;
    gemm_bf16_wmma<<<cdiv(N, 32) * tn, 32, 0, stream>>>(h2, W2, g2, N, 64, 32, tn);
  }
  attn_scores1<<<cdiv(N, TB), TB, 0, stream>>>(g2, aS2w, aD2w, aS2, aD2, N);
  fill_f<<<cdiv(N, TB), TB, 0, stream>>>(mx2, -INFINITY, N);
  fill_f<<<cdiv(N, TB), TB, 0, stream>>>(dn2, 0.f, N);
  fill_bias<<<cdiv((long long)N * 32, TB), TB, 0, stream>>>(out, b2, 31, N * 32);
  edge_max1<<<cdiv(ET, TB), TB, 0, stream>>>(ei, aS2, aD2, mx2, E, ET);
  edge_den1<<<cdiv(ET, TB), TB, 0, stream>>>(ei, aS2, aD2, mx2, dn2, E, ET);
  edge_agg1<<<cdiv((long long)ET * 4, TB), TB, 0, stream>>>(ei, aS2, aD2, mx2, dn2, g2, out, E, ET);
}
